// MADESplit_1125281432154
// MI455X (gfx1250) — matrix operand order repaired and checked by reading, not compiled, not run on hardware
//
#include <hip/hip_runtime.h>
#include <hip/hip_bf16.h>

typedef __bf16 bf16_t;
typedef __attribute__((ext_vector_type(16))) __bf16 v16bf;
typedef __attribute__((ext_vector_type(8)))  __bf16 v8bf;
typedef __attribute__((ext_vector_type(8)))  float  v8f;

// ---------------------------------------------------------------------------
// Prep: apply MADE mask analytically and convert f32 -> bf16.
// mtype 0: input mask  [H,D]: keep = (r%127) >= c
// mtype 1: hidden mask [H,H]: keep = (r%127) >= (c%127)
// mtype 2: output mask [D,H]: keep = (r-1)   >= (c%127)
// mtype 3: plain convert (no mask)
// ---------------------------------------------------------------------------
__global__ void mask_cvt_kernel(const float* __restrict__ src,
                                bf16_t* __restrict__ dst,
                                int rows, int cols, int mtype) {
    int idx = blockIdx.x * blockDim.x + threadIdx.x;
    if (idx >= rows * cols) return;
    int r = idx / cols;
    int c = idx - r * cols;
    bool keep = true;
    if      (mtype == 0) keep = (r % 127) >= c;
    else if (mtype == 1) keep = (r % 127) >= (c % 127);
    else if (mtype == 2) keep = (r - 1)   >= (c % 127);
    float v = keep ? src[idx] : 0.0f;
    dst[idx] = (bf16_t)v;
}

// ---------------------------------------------------------------------------
// Tiled WMMA GEMM: C[M,N] = act(A[M,K] * W[N,K]^T + bias[N])
// A, W in bf16 row-major. Block tile 128x128, K-step 32.
// 8 waves/block, each wave owns a 32x64 tile = 2x4 wmma accumulators.
// ACT: 0=none, 1=tanh, 2=relu.  OUT_BF16: 1 -> bf16 out, 0 -> f32 out.
// ---------------------------------------------------------------------------
#define MBLK 128
#define NBLK 128
#define KBLK 32
#define LDSS 40   // bf16 elems per LDS row: 32 + 8 pad (16B aligned rows)

template<int ACT, int OUT_BF16>
__global__ __launch_bounds__(256)
void gemm_wmma_kernel(const bf16_t* __restrict__ A,
                      const bf16_t* __restrict__ W,
                      const float*  __restrict__ bias,
                      void* __restrict__ Cout,
                      int M, int N, int K) {
    __shared__ bf16_t lA[MBLK * LDSS];
    __shared__ bf16_t lB[NBLK * LDSS];

    const int tid    = threadIdx.x;
    const int lane   = tid & 31;
    const int wave   = tid >> 5;
    const int laneLo = lane & 15;
    const int laneHi = lane >> 4;          // 0 or 1

    const int blockM = blockIdx.x * MBLK;
    const int blockN = blockIdx.y * NBLK;

    // wave tile: 4 waves along M (32 rows each), 2 waves along N (64 cols each)
    const int waveM = (wave & 3) * 32;
    const int waveN = (wave >> 2) * 64;

    v8f zero = {};
    v8f acc[2][4];
#pragma unroll
    for (int i = 0; i < 2; ++i)
#pragma unroll
        for (int j = 0; j < 4; ++j)
            acc[i][j] = zero;

    // cooperative staging: 256 threads, each moves 32B of A and 32B of B
    const int ldRow = tid >> 1;            // 0..127
    const int ldCol = (tid & 1) * 16;      // 0 or 16 (bf16 elems)

    for (int k0 = 0; k0 < K; k0 += KBLK) {
        const bf16_t* gA = A + (size_t)(blockM + ldRow) * K + k0 + ldCol;
        const bf16_t* gB = W + (size_t)(blockN + ldRow) * K + k0 + ldCol;
        v8bf a0 = *(const v8bf*)(gA);
        v8bf a1 = *(const v8bf*)(gA + 8);
        v8bf b0 = *(const v8bf*)(gB);
        v8bf b1 = *(const v8bf*)(gB + 8);
        if (k0 + KBLK < K) {
            __builtin_prefetch(gA + KBLK, 0, 1);   // global_prefetch_b8
            __builtin_prefetch(gB + KBLK, 0, 1);
        }
        __syncthreads();                   // prior iteration done reading LDS
        *(v8bf*)&lA[ldRow * LDSS + ldCol]     = a0;
        *(v8bf*)&lA[ldRow * LDSS + ldCol + 8] = a1;
        *(v8bf*)&lB[ldRow * LDSS + ldCol]     = b0;
        *(v8bf*)&lB[ldRow * LDSS + ldCol + 8] = b1;
        __syncthreads();

        // Fragment gathers per ISA VGPR layouts (wave32).
        // A 16x32 bf16: lanes 0-15 -> M=lane, K 0..7 (V0-3) & 16..23 (V4-7);
        //               lanes 16-31 -> same M, K 8..15 & 24..31.
        union { v16bf v; v8bf h[2]; } afr[2], bfr[4];
#pragma unroll
        for (int i = 0; i < 2; ++i) {
            const bf16_t* p = &lA[(waveM + i * 16 + laneLo) * LDSS + laneHi * 8];
            afr[i].h[0] = *(const v8bf*)(p);
            afr[i].h[1] = *(const v8bf*)(p + 16);
        }
        // B 32x16: lanes 0-15 -> col=lane, K 0..7 & 16..23; lanes 16-31 -> K 8..15 & 24..31.
#pragma unroll
        for (int j = 0; j < 4; ++j) {
            const bf16_t* p = &lB[(waveN + j * 16 + laneLo) * LDSS + laneHi * 8];
            bfr[j].h[0] = *(const v8bf*)(p);
            bfr[j].h[1] = *(const v8bf*)(p + 16);
        }
#pragma unroll
        for (int i = 0; i < 2; ++i)
#pragma unroll
            for (int j = 0; j < 4; ++j)
                acc[i][j] = __builtin_amdgcn_wmma_f32_16x16x32_bf16(
                    false, afr[i].v, false, bfr[j].v,
                    (short)0, acc[i][j], false, false);
    }

    // epilogue: C layout -> lane laneLo = col, laneHi selects rows 8..15,
    // vgpr v = row within half-tile.
#pragma unroll
    for (int j = 0; j < 4; ++j) {
        const int n = blockN + waveN + j * 16 + laneLo;
        const float bv = bias[n];
#pragma unroll
        for (int i = 0; i < 2; ++i) {
            const int mBase = blockM + waveM + i * 16 + laneHi * 8;
#pragma unroll
            for (int v = 0; v < 8; ++v) {
                float val = acc[i][j][v] + bv;
                if      (ACT == 1) val = tanhf(val);
                else if (ACT == 2) val = fmaxf(val, 0.0f);
                const size_t off = (size_t)(mBase + v) * N + n;
                if (OUT_BF16) ((bf16_t*)Cout)[off] = (bf16_t)val;
                else          ((float*)Cout)[off]  = val;
            }
        }
    }
}

// ---------------------------------------------------------------------------
// Final combine: u = (x - m) * exp(-a); out2[row] = -sum_d a[row,d]
// ---------------------------------------------------------------------------
__global__ void final_kernel(const float* __restrict__ X,
                             const float* __restrict__ m,
                             const float* __restrict__ a,
                             float* __restrict__ out,
                             int B, int D) {
    const int row = blockIdx.x;
    const int d   = threadIdx.x;           // blockDim.x == D == 128
    const size_t idx = (size_t)row * D + d;
    const float av = a[idx];
    out[idx] = (X[idx] - m[idx]) * expf(-av);

    float s = av;
#pragma unroll
    for (int off = 16; off; off >>= 1) s += __shfl_xor(s, off);

    __shared__ float part[4];
    if ((d & 31) == 0) part[d >> 5] = s;
    __syncthreads();
    if (d == 0)
        out[(size_t)B * D + row] = -(part[0] + part[1] + part[2] + part[3]);
}

// ---------------------------------------------------------------------------
extern "C" void kernel_launch(void* const* d_in, const int* in_sizes, int n_in,
                              void* d_out, int out_size, void* d_ws, size_t ws_size,
                              hipStream_t stream) {
    (void)in_sizes; (void)n_in; (void)out_size; (void)ws_size;

    const int B = 8192, D = 128, H = 2048;

    const float* X    = (const float*)d_in[0];
    const float* s_w0 = (const float*)d_in[1];
    const float* s_b0 = (const float*)d_in[2];
    const float* s_w1 = (const float*)d_in[3];
    const float* s_b1 = (const float*)d_in[4];
    const float* s_w2 = (const float*)d_in[5];
    const float* s_b2 = (const float*)d_in[6];
    const float* t_w0 = (const float*)d_in[7];
    const float* t_b0 = (const float*)d_in[8];
    const float* t_w1 = (const float*)d_in[9];
    const float* t_b1 = (const float*)d_in[10];
    const float* t_w2 = (const float*)d_in[11];
    const float* t_b2 = (const float*)d_in[12];

    char* ws = (char*)d_ws;
    size_t off = 0;
    auto alloc = [&](size_t bytes) -> void* {
        void* p = ws + off;
        off = (off + bytes + 255) & ~(size_t)255;
        return p;
    };

    bf16_t* xbf = (bf16_t*)alloc((size_t)B * D * 2);
    bf16_t* w0s = (bf16_t*)alloc((size_t)H * D * 2);
    bf16_t* w1s = (bf16_t*)alloc((size_t)H * H * 2);
    bf16_t* w2s = (bf16_t*)alloc((size_t)D * H * 2);
    bf16_t* w0t = (bf16_t*)alloc((size_t)H * D * 2);
    bf16_t* w1t = (bf16_t*)alloc((size_t)H * H * 2);
    bf16_t* w2t = (bf16_t*)alloc((size_t)D * H * 2);
    bf16_t* h0  = (bf16_t*)alloc((size_t)B * H * 2);
    bf16_t* h1  = (bf16_t*)alloc((size_t)B * H * 2);
    float*  mb  = (float*) alloc((size_t)B * D * 4);
    float*  ab  = (float*) alloc((size_t)B * D * 4);

    auto cvt = [&](const float* src, bf16_t* dst, int rows, int cols, int mtype) {
        int n = rows * cols;
        mask_cvt_kernel<<<(n + 255) / 256, 256, 0, stream>>>(src, dst, rows, cols, mtype);
    };

    // prep: masked bf16 weights + bf16 input
    cvt(X,    xbf, B, D, 3);
    cvt(s_w0, w0s, H, D, 0);
    cvt(s_w1, w1s, H, H, 1);
    cvt(s_w2, w2s, D, H, 2);
    cvt(t_w0, w0t, H, D, 0);
    cvt(t_w1, w1t, H, H, 1);
    cvt(t_w2, w2t, D, H, 2);

    dim3 blk(256);
    dim3 gBH(B / MBLK, H / NBLK);   // [B,H] outputs
    dim3 gBD(B / MBLK, D / NBLK);   // [B,D] outputs

    // s-net (tanh)
    gemm_wmma_kernel<1, 1><<<gBH, blk, 0, stream>>>(xbf, w0s, s_b0, h0, B, H, D);
    gemm_wmma_kernel<1, 1><<<gBH, blk, 0, stream>>>(h0,  w1s, s_b1, h1, B, H, H);
    gemm_wmma_kernel<0, 0><<<gBD, blk, 0, stream>>>(h1,  w2s, s_b2, mb, B, D, H);

    // t-net (relu)
    gemm_wmma_kernel<2, 1><<<gBH, blk, 0, stream>>>(xbf, w0t, t_b0, h0, B, H, D);
    gemm_wmma_kernel<2, 1><<<gBH, blk, 0, stream>>>(h0,  w1t, t_b1, h1, B, H, H);
    gemm_wmma_kernel<0, 0><<<gBD, blk, 0, stream>>>(h1,  w2t, t_b2, ab, B, D, H);

    // combine
    final_kernel<<<B, D, 0, stream>>>(X, mb, ab, (float*)d_out, B, D);
}
